// MultiHeadAttention_39290360824518
// MI455X (gfx1250) — matrix operand order repaired and checked by reading, not compiled, not run on hardware
//
#include <hip/hip_runtime.h>
#include <hip/hip_bf16.h>

// ---------------------------------------------------------------------------
// MHA forward for B=4, S=2048, D=1024, H=16, DK=DV=64 on gfx1250 (wave32).
// All matmuls via v_wmma_f32_16x16x32_bf16 (fp32 accumulate).
// Attention K tiles staged via the Tensor Data Mover (tensor_load_to_lds).
// ---------------------------------------------------------------------------

typedef __attribute__((ext_vector_type(8)))  __bf16 v8bf;
typedef __attribute__((ext_vector_type(16))) __bf16 v16bf;
typedef __attribute__((ext_vector_type(8)))  float  v8f;
typedef __attribute__((ext_vector_type(4)))  unsigned int uint32x4;
typedef __attribute__((ext_vector_type(8)))  int int32x8;
typedef __attribute__((ext_vector_type(4)))  int int32x4;

__device__ __forceinline__ v16bf ld16(const __bf16* p0, const __bf16* p1) {
  v8bf a = *(const v8bf*)p0;
  v8bf b = *(const v8bf*)p1;
  return __builtin_shufflevector(a, b, 0,1,2,3,4,5,6,7,8,9,10,11,12,13,14,15);
}

__device__ __forceinline__ v8f wmma_bf16(v16bf a, v16bf b, v8f c) {
  return __builtin_amdgcn_wmma_f32_16x16x32_bf16(false, a, false, b, (short)0, c, false, false);
}

// TDM: DMA a 64x64 bf16 tile (row-major, row stride 64 elements) from global
// into LDS at lds_off. 2-D descriptor per CDNA5 ISA.
// 6-arg builtin form (clang-23): (g0, g1, g2, g3, g4, cpol); groups 2+ unused
// for a 2-D tensor -> zero-filled.
__device__ __forceinline__ void tdm_load_tile64x64_bf16(const __bf16* gptr,
                                                        unsigned lds_off) {
  unsigned long long ga = (unsigned long long)(size_t)gptr;
  uint32x4 g0;
  g0[0] = 1u;                                         // count=1, user flags 0
  g0[1] = lds_off;                                    // lds_addr
  g0[2] = (unsigned)(ga & 0xffffffffu);               // global_addr[31:0]
  g0[3] = (unsigned)((ga >> 32) & 0x01ffffffu)        // global_addr[56:32]
        | (2u << 30);                                 // type=2 (image)
  int32x8 g1;
  g1[0] = 1 << 16;          // workgroup_mask=0, data_size=1 (2 bytes)
  g1[1] = 64 << 16;         // tensor_dim0[15:0] = 64
  g1[2] = 64 << 16;         // tensor_dim0[31:16]=0, tensor_dim1[15:0] = 64
  g1[3] = 64 << 16;         // tensor_dim1[31:16]=0, tile_dim0 = 64
  g1[4] = 64;               // tile_dim1 = 64, tile_dim2 = 0
  g1[5] = 64;               // tensor_dim0_stride[31:0] = 64 elements
  g1[6] = 0;                // dim0_stride[47:32]=0, dim1_stride[15:0]=0
  g1[7] = 0;
  int32x4 z4 = {0, 0, 0, 0};
  int32x8 z8 = {0, 0, 0, 0, 0, 0, 0, 0};
  __builtin_amdgcn_tensor_load_to_lds(g0, g1, z4, z4, z8, 0);
}

// ---------------------------------------------------------------------------
// GEMM: C[M,N] = A[M,K] (fp32) * W[K,N] (fp32) + bias, computed in bf16 WMMA.
// MODE 0: output -> bf16 repacked as [B][H][S][64]  (QKV staging)
// MODE 1: output -> fp32 row-major [M,N]            (final projection)
// Block tile 128x128, BK=64, 8 waves (4 along M x 2 along N), wave tile 32x64.
// ---------------------------------------------------------------------------
#define BM 128
#define BN 128
#define BK 64
#define LDT_S 72   // bf16 elements; 144B row stride (16B aligned)

template <int MODE>
__global__ __launch_bounds__(256)
void gemm_bf16_wmma(const float* __restrict__ A, const float* __restrict__ W,
                    const float* __restrict__ bias, void* __restrict__ out,
                    int M, int K, int N)
{
  __shared__ __bf16 As[BM * LDT_S];      // [m][k]
  __shared__ __bf16 Bs[BN * LDT_S];      // [n][k]  (transposed for B-fragments)

  const int tid  = threadIdx.x;
  const int lane = tid & 31;
  const int wave = tid >> 5;
  const int lh   = lane >> 4;            // half-wave id (K/M split per ISA layout)
  const int ll   = lane & 15;
  const int m0   = blockIdx.y * BM;
  const int n0   = blockIdx.x * BN;
  const int wm   = (wave >> 1) * 32;     // wave offset along M
  const int wn   = (wave & 1) * 64;      // wave offset along N

  v8f acc[2][4] = {};

  for (int k0 = 0; k0 < K; k0 += BK) {
    __syncthreads();
    // Stage A tile (128x64 fp32 -> bf16). 2048 float4 / 256 threads.
#pragma unroll
    for (int i = 0; i < 8; ++i) {
      int f   = tid + i * 256;
      int row = f >> 4;
      int kq  = (f & 15) << 2;
      const float* src = A + (size_t)(m0 + row) * K + k0 + kq;
      const float4 v = *(const float4*)src;
      if (k0 + BK < K) __builtin_prefetch(src + BK, 0, 1);
      __bf16* d = &As[row * LDT_S + kq];
      d[0] = (__bf16)v.x; d[1] = (__bf16)v.y; d[2] = (__bf16)v.z; d[3] = (__bf16)v.w;
    }
    // Stage B tile (64x128 fp32), transposed into Bs[n][k].
#pragma unroll
    for (int i = 0; i < 8; ++i) {
      int f  = tid + i * 256;
      int kk = f >> 5;
      int nq = (f & 31) << 2;
      const float* src = W + (size_t)(k0 + kk) * N + n0 + nq;
      const float4 v = *(const float4*)src;
      if (k0 + BK < K) __builtin_prefetch(src + (size_t)BK * N, 0, 1);
      Bs[(nq + 0) * LDT_S + kk] = (__bf16)v.x;
      Bs[(nq + 1) * LDT_S + kk] = (__bf16)v.y;
      Bs[(nq + 2) * LDT_S + kk] = (__bf16)v.z;
      Bs[(nq + 3) * LDT_S + kk] = (__bf16)v.w;
    }
    __syncthreads();

#pragma unroll
    for (int ks = 0; ks < 2; ++ks) {
      // A fragments: lane ll = row, K chunks at lh*8 and lh*8+16.
      v16bf af[2];
#pragma unroll
      for (int mi = 0; mi < 2; ++mi) {
        const __bf16* p = &As[(wm + mi * 16 + ll) * LDT_S + ks * 32 + lh * 8];
        af[mi] = ld16(p, p + 16);
      }
      // B fragments: lane ll = column, K chunks at lh*8 and lh*8+16.
#pragma unroll
      for (int ni = 0; ni < 4; ++ni) {
        const __bf16* p = &Bs[(wn + ni * 16 + ll) * LDT_S + ks * 32 + lh * 8];
        v16bf bf = ld16(p, p + 16);
#pragma unroll
        for (int mi = 0; mi < 2; ++mi)
          acc[mi][ni] = wmma_bf16(af[mi], bf, acc[mi][ni]);
      }
    }
  }

  // Epilogue: C-layout element (r, lane) -> row = r + lh*8, col = ll.
#pragma unroll
  for (int mi = 0; mi < 2; ++mi)
#pragma unroll
    for (int ni = 0; ni < 4; ++ni) {
      int col = n0 + wn + ni * 16 + ll;
      float bv = bias[col];
#pragma unroll
      for (int r = 0; r < 8; ++r) {
        int row = m0 + wm + mi * 16 + r + lh * 8;
        float val = acc[mi][ni][r] + bv;
        if (MODE == 0) {
          int b = row >> 11, s = row & 2047;     // S = 2048
          int h = col >> 6,  dk = col & 63;      // head dim 64
          ((__bf16*)out)[((((size_t)b * 16 + h) << 11) + s) * 64 + dk] = (__bf16)val;
        } else {
          ((float*)out)[(size_t)row * N + col] = val;
        }
      }
    }
}

// ---------------------------------------------------------------------------
// Flash attention: per block = (128 query rows, one head). 8 waves, each wave
// owns 16 query rows. KV tiled by 64 with online softmax. K tile staged via
// TDM (tensor_load_to_lds); V tile staged with an LDS transpose. Outputs the
// reference "weights" tensor (B,S,H*64) in fp32.
// ---------------------------------------------------------------------------
__global__ __launch_bounds__(256)
void attn_wmma(const __bf16* __restrict__ Qb, const __bf16* __restrict__ Kb,
               const __bf16* __restrict__ Vb, const float* __restrict__ mask,
               float* __restrict__ Wout)
{
  __shared__ __bf16 Ks[64 * 64];          // [kv][dk] row-major (TDM target)
  __shared__ __bf16 Vs[64 * 72];          // [dv][kv] transposed V tile
  __shared__ __bf16 Ps[8 * 16 * 72];      // per-wave P staging [wave][row][kv]

  const int tid  = threadIdx.x;
  const int lane = tid & 31;
  const int wave = tid >> 5;
  const int lh   = lane >> 4;
  const int ll   = lane & 15;
  const int qb   = blockIdx.x;            // 0..15  query block
  const int h    = blockIdx.y;            // 0..15  head
  const int b    = blockIdx.z;            // 0..3   batch

  const size_t headoff = (((size_t)b * 16 + h) << 11) * 64;   // * S * 64
  const __bf16* Qh = Qb + headoff;
  const __bf16* Kh = Kb + headoff;
  const __bf16* Vh = Vb + headoff;
  const int q0 = qb * 128 + wave * 16;
  const unsigned ksOff = (unsigned)(size_t)(void*)&Ks[0];

  // Q fragments for rows q0..q0+15, dk 0..63 (two K=32 chunks).
  v16bf qf[2];
  {
    const __bf16* qp = Qh + (size_t)(q0 + ll) * 64;
    qf[0] = ld16(qp + lh * 8,      qp + lh * 8 + 16);
    qf[1] = ld16(qp + 32 + lh * 8, qp + 32 + lh * 8 + 16);
  }

  v8f o[4] = {};
  float mrow[8], lrow[8];
#pragma unroll
  for (int r = 0; r < 8; ++r) { mrow[r] = -1e30f; lrow[r] = 0.0f; }

  const float scale = 0.125f;             // 1/sqrt(64)
  __bf16* Pw = &Ps[wave * 16 * 72];

  for (int kv0 = 0; kv0 < 2048; kv0 += 64) {
    __syncthreads();                      // prior reads of Ks/Vs complete

    // Async K tile DMA: one wave issues, TENSORcnt tracked, barrier publishes.
    if (wave == 0) tdm_load_tile64x64_bf16(Kh + (size_t)kv0 * 64, ksOff);

    // Stage V tile transposed: Vs[dv][kv] = V[kv0+kv][dv]. 512 x 8-bf16 chunks.
#pragma unroll
    for (int i = 0; i < 2; ++i) {
      int e  = tid + i * 256;
      int kv = e >> 3;
      int dv = (e & 7) << 3;
      const __bf16* src = Vh + (size_t)(kv0 + kv) * 64 + dv;
      v8bf v = *(const v8bf*)src;
      if (kv0 + 64 < 2048) __builtin_prefetch(src + 64 * 64, 0, 1);
#pragma unroll
      for (int j = 0; j < 8; ++j) Vs[(dv + j) * 72 + kv] = v[j];
    }

    if (wave == 0) __builtin_amdgcn_s_wait_tensorcnt(0);
    __syncthreads();

    // Scores: S = Q(16x64) * K^T(64x64) -> 4 accumulators of 16x16.
    // Ks rows are B-fragment rows (lane = kv column, dk chunks at lh*8 and lh*8+16).
    v8f sc[4] = {};
#pragma unroll
    for (int a = 0; a < 4; ++a) {
      const __bf16* kp = &Ks[(a * 16 + ll) * 64];
      v16bf b0 = ld16(kp + lh * 8,      kp + lh * 8 + 16);
      v16bf b1 = ld16(kp + 32 + lh * 8, kp + 32 + lh * 8 + 16);
      sc[a] = wmma_bf16(qf[0], b0, sc[a]);
      sc[a] = wmma_bf16(qf[1], b1, sc[a]);
    }

    // scale + additive mask
#pragma unroll
    for (int a = 0; a < 4; ++a) {
      float mv = mask[(size_t)b * 2048 + kv0 + a * 16 + ll];
#pragma unroll
      for (int r = 0; r < 8; ++r) sc[a][r] = sc[a][r] * scale + mv;
    }

    // Online softmax per row (rows live in 16-lane halves per C layout).
#pragma unroll
    for (int r = 0; r < 8; ++r) {
      float mx = fmaxf(fmaxf(sc[0][r], sc[1][r]), fmaxf(sc[2][r], sc[3][r]));
#pragma unroll
      for (int d = 1; d < 16; d <<= 1) mx = fmaxf(mx, __shfl_xor(mx, d, 32));
      float mnew = fmaxf(mrow[r], mx);
      float corr = __expf(mrow[r] - mnew);
      float rs = 0.0f;
#pragma unroll
      for (int a = 0; a < 4; ++a) {
        float p = __expf(sc[a][r] - mnew);
        sc[a][r] = p;
        rs += p;
      }
#pragma unroll
      for (int d = 1; d < 16; d <<= 1) rs += __shfl_xor(rs, d, 32);
      lrow[r] = lrow[r] * corr + rs;
      mrow[r] = mnew;
#pragma unroll
      for (int a = 0; a < 4; ++a) o[a][r] *= corr;
    }

    // Stage P (C layout -> A layout) through per-wave LDS as bf16.
#pragma unroll
    for (int a = 0; a < 4; ++a)
#pragma unroll
      for (int r = 0; r < 8; ++r)
        Pw[(r + lh * 8) * 72 + a * 16 + ll] = (__bf16)sc[a][r];

    // O += P(16x64) * V(64x64)
#pragma unroll
    for (int c = 0; c < 2; ++c) {
      const __bf16* pp = &Pw[ll * 72 + c * 32 + lh * 8];
      v16bf pf = ld16(pp, pp + 16);
#pragma unroll
      for (int a = 0; a < 4; ++a) {
        const __bf16* vp = &Vs[(a * 16 + ll) * 72 + c * 32 + lh * 8];
        v16bf vf = ld16(vp, vp + 16);
        o[a] = wmma_bf16(pf, vf, o[a]);
      }
    }
  }

  // Normalize and emit weights (B,S,H*64) fp32.
#pragma unroll
  for (int r = 0; r < 8; ++r) {
    float inv = 1.0f / lrow[r];
    int s = q0 + r + lh * 8;
#pragma unroll
    for (int a = 0; a < 4; ++a) {
      int col = h * 64 + a * 16 + ll;
      Wout[((size_t)b * 2048 + s) * 1024 + col] = o[a][r] * inv;
    }
  }
}

// ---------------------------------------------------------------------------
extern "C" void kernel_launch(void* const* d_in, const int* in_sizes, int n_in,
                              void* d_out, int out_size, void* d_ws, size_t ws_size,
                              hipStream_t stream)
{
  const float* query = (const float*)d_in[0];
  const float* key   = (const float*)d_in[1];
  const float* value = (const float*)d_in[2];
  const float* mask  = (const float*)d_in[3];
  const float* Wq = (const float*)d_in[4];  const float* bq = (const float*)d_in[5];
  const float* Wk = (const float*)d_in[6];  const float* bk = (const float*)d_in[7];
  const float* Wv = (const float*)d_in[8];  const float* bv = (const float*)d_in[9];
  const float* Wo = (const float*)d_in[10]; const float* bo = (const float*)d_in[11];

  const int M = 8192, K = 1024, N = 1024;          // B*S, D, H*dk
  const size_t headElems = (size_t)4 * 16 * 2048 * 64;   // 8 Mi elements

  __bf16* qbuf = (__bf16*)d_ws;
  __bf16* kbuf = qbuf + headElems;
  __bf16* vbuf = kbuf + headElems;

  float* out_proj = (float*)d_out;                       // (B,S,D)
  float* weights  = out_proj + (size_t)M * 1024;         // (B,S,H*DV)

  dim3 gemmGrid(N / BN, M / BM);                         // (8, 64)

  // QKV projections -> bf16 head-major staging
  gemm_bf16_wmma<0><<<gemmGrid, 256, 0, stream>>>(query, Wq, bq, qbuf, M, K, N);
  gemm_bf16_wmma<0><<<gemmGrid, 256, 0, stream>>>(key,   Wk, bk, kbuf, M, K, N);
  gemm_bf16_wmma<0><<<gemmGrid, 256, 0, stream>>>(value, Wv, bv, vbuf, M, K, N);

  // Attention -> weights (fp32)
  attn_wmma<<<dim3(16, 16, 4), 256, 0, stream>>>(qbuf, kbuf, vbuf, mask, weights);

  // Output projection: output = weights @ Wo + bo
  gemm_bf16_wmma<1><<<gemmGrid, 256, 0, stream>>>(weights, Wo, bo, out_proj, M, K, N);
}
